// VMamba_35622458753231
// MI455X (gfx1250) — matrix operand order repaired and checked by reading, not compiled, not run on hardware
//
#include <hip/hip_runtime.h>
#include <hip/hip_bf16.h>

typedef __attribute__((ext_vector_type(16))) _Float16 v16h;
typedef __attribute__((ext_vector_type(8)))  _Float16 v8h;
typedef __attribute__((ext_vector_type(4)))  _Float16 v4h;
typedef __attribute__((ext_vector_type(8)))  float    v8f;

#define BATCH 4
#define LSEQ  1024
#define BL    4096   // BATCH*LSEQ
#define NSTATE 16

// ---------------------------------------------------------------------------
// Epilogue modes for the WMMA GEMM
// ---------------------------------------------------------------------------
#define EPI_NONE     0
#define EPI_BIAS     1
#define EPI_BIAS_SP  2   // bias + softplus  (Mamba delta)
#define EPI_BIAS_RES 3   // bias + residual  (block merge)

__device__ __forceinline__ float softplus_f(float x) {
  return (x > 20.f) ? x : __logf(1.f + __expf(x));
}
__device__ __forceinline__ float silu_f(float x) {
  return x / (1.f + __expf(-x));
}

// A fragment: 16x32 f16, M rows on lanes 0-15 (mirrored on 16-31), per ISA:
//   lane<16 : half i<8 -> K=kb+i,  half i>=8 -> K=kb+8+i   (kb=k0)
//   lane>=16: same with kb=k0+8
__device__ __forceinline__ v16h load_a_frag(const float* __restrict__ A,
                                            int lda, int row, int kb) {
  const float* p = A + (size_t)row * lda + kb;
  float4 q0 = *(const float4*)(p);
  float4 q1 = *(const float4*)(p + 4);
  float4 q2 = *(const float4*)(p + 16);
  float4 q3 = *(const float4*)(p + 20);
  v16h a;
  a[0]=(_Float16)q0.x; a[1]=(_Float16)q0.y; a[2]=(_Float16)q0.z; a[3]=(_Float16)q0.w;
  a[4]=(_Float16)q1.x; a[5]=(_Float16)q1.y; a[6]=(_Float16)q1.z; a[7]=(_Float16)q1.w;
  a[8]=(_Float16)q2.x; a[9]=(_Float16)q2.y; a[10]=(_Float16)q2.z; a[11]=(_Float16)q2.w;
  a[12]=(_Float16)q3.x; a[13]=(_Float16)q3.y; a[14]=(_Float16)q3.z; a[15]=(_Float16)q3.w;
  return a;
}

// D = A(4096,K) * B(K,N), f16 WMMA with f32 accum. M fixed = 4096.
// Block: 256 threads = 8 waves; wave w handles rows [bx*128+16w, +16).
// NT (compile-time) 16-wide N-subtiles per wave; grid.y covers N in 16*NT chunks.
// N must be a multiple of 16*NT; K a multiple of 32.
// B tile is staged through LDS in WMMA-fragment order:
//   smem[(j*32 + lane)*16 + i] = f16( B[k0 + (lane>=16?8:0) + (i<8?i:8+i)][nbase + j*16 + (lane&15)] )
template <int EPI, int NT>
__global__ void gemm_wmma(const float* __restrict__ A, const float* __restrict__ B,
                          float* __restrict__ D, const float* __restrict__ bias,
                          const float* __restrict__ R,
                          int N, int K, int lda, int ldb, int ldd, int colbase, int ldr) {
  constexpr int NW = 16 * NT;        // block tile width in N
  constexpr int KC = 32 * NW / (256 * 2); // halved twice below; recompute cleanly:
  constexpr int GR = 256 / NW;       // k-groups of staging threads
  constexpr int KCH = 32 / GR;       // consecutive k's staged per thread (8 or 4)
  (void)KC;

  __shared__ __align__(16) _Float16 smem[4 * 32 * 16];  // up to 4 subtiles

  const int lane  = threadIdx.x & 31;
  const int wave  = threadIdx.x >> 5;
  const int m0    = blockIdx.x * 128 + wave * 16;
  const int nbase = blockIdx.y * NW;

  // staging coordinates for this thread
  const int nn  = threadIdx.x % NW;
  const int g   = threadIdx.x / NW;
  const int kk0 = g * KCH;                         // never straddles k=16
  const int sj  = nn >> 4;
  const int sl  = (nn & 15) + (((kk0 >> 3) & 1) ? 16 : 0);
  const int si0 = (kk0 & 7) + ((kk0 >= 16) ? 8 : 0);
  _Float16* sptr = &smem[(sj * 32 + sl) * 16 + si0];

  v8f acc[NT];
#pragma unroll
  for (int j = 0; j < NT; ++j)
#pragma unroll
    for (int r = 0; r < 8; ++r) acc[j][r] = 0.f;

  const int arow = m0 + (lane & 15);
  for (int k0 = 0; k0 < K; k0 += 32) {
    __syncthreads();                               // previous reads done
    const float* bp = B + (size_t)(k0 + kk0) * ldb + nbase + nn;
    if constexpr (KCH == 8) {
      v8h t;
#pragma unroll
      for (int q = 0; q < 8; ++q) t[q] = (_Float16)bp[(size_t)q * ldb];
      *(v8h*)sptr = t;
    } else {
      v4h t;
#pragma unroll
      for (int q = 0; q < 4; ++q) t[q] = (_Float16)bp[(size_t)q * ldb];
      *(v4h*)sptr = t;
    }
    __syncthreads();                               // tile visible

    const int kb = k0 + ((lane >> 4) << 3);
    v16h afrag = load_a_frag(A, lda, arow, kb);
    v16h bfrag[NT];
#pragma unroll
    for (int j = 0; j < NT; ++j)
      bfrag[j] = *(const v16h*)&smem[(j * 32 + lane) * 16];
#pragma unroll
    for (int j = 0; j < NT; ++j)
      acc[j] = __builtin_amdgcn_wmma_f32_16x16x32_f16(
          false, afrag, false, bfrag[j], (short)0, acc[j], false, false);
  }

  const int rowg = m0 + ((lane >> 4) << 3);        // C/D: vgpr j -> row rowg+j
#pragma unroll
  for (int j = 0; j < NT; ++j) {
    const int col = nbase + j * 16 + (lane & 15);
    float bv = 0.f;
    if constexpr (EPI != EPI_NONE) bv = bias[col];
#pragma unroll
    for (int r = 0; r < 8; ++r) {
      float v = acc[j][r] + bv;
      if constexpr (EPI == EPI_BIAS_SP)  v = softplus_f(v);
      if constexpr (EPI == EPI_BIAS_RES) v += R[(size_t)(rowg + r) * ldr + col];
      D[(size_t)(rowg + r) * ldd + colbase + col] = v;
    }
  }
}

// ---------------------------------------------------------------------------
// LayerNorm: one wave32 per row.
// ---------------------------------------------------------------------------
__global__ void layernorm_rows(const float* __restrict__ X, const float* __restrict__ w,
                               const float* __restrict__ b, float* __restrict__ Y,
                               int rows, int C) {
  const int gw   = (blockIdx.x * blockDim.x + threadIdx.x) >> 5;
  const int lane = threadIdx.x & 31;
  if (gw >= rows) return;
  const float* x = X + (size_t)gw * C;
  float s = 0.f, ss = 0.f;
  for (int c = lane; c < C; c += 32) { float v = x[c]; s += v; ss += v * v; }
#pragma unroll
  for (int off = 16; off; off >>= 1) {
    s  += __shfl_xor(s,  off, 32);
    ss += __shfl_xor(ss, off, 32);
  }
  const float mu  = s / (float)C;
  const float var = ss / (float)C - mu * mu;
  const float inv = rsqrtf(var + 1e-5f);
  float* y = Y + (size_t)gw * C;
  for (int c = lane; c < C; c += 32) y[c] = (x[c] - mu) * inv * w[c] + b[c];
}

// seq_v[b, w*32+h, c] = seq_h[b, h*32+w, c]
__global__ void transpose_tok(const float* __restrict__ in, float* __restrict__ out, int C) {
  const int total = BL * C;
  int idx = blockIdx.x * blockDim.x + threadIdx.x;
  if (idx >= total) return;
  int c = idx % C;
  int l = (idx / C) & (LSEQ - 1);
  int b = idx / (C * LSEQ);
  int h = l >> 5, w2 = l & 31;
  out[((size_t)(b * LSEQ + w2 * 32 + h)) * C + c] = in[idx];
}

// depthwise conv3 (pad 1, per-sequence) + bias + SiLU on xc = xz[:, :di]
__global__ void dwconv_silu(const float* __restrict__ xz, const float* __restrict__ cw,
                            const float* __restrict__ cb, float* __restrict__ xc, int di) {
  const int total = BL * di;
  int idx = blockIdx.x * blockDim.x + threadIdx.x;
  if (idx >= total) return;
  int d   = idx % di;
  int row = idx / di;
  int l   = row & (LSEQ - 1);
  const float* base = xz + (size_t)row * (2 * di) + d;
  float acc = cb[d];
  const float w0 = cw[d], w1 = cw[di + d], w2 = cw[2 * di + d];
  if (l > 0)        acc += base[-(ptrdiff_t)(2 * di)] * w0;
  acc += base[0] * w1;
  if (l < LSEQ - 1) acc += base[2 * di] * w2;
  xc[idx] = silu_f(acc);
}

// selective scan: one lane per (batch, channel); 16 states in registers.
// g = (scan_out + D*xc) * silu(z),  z = xz[:, di + d]
__global__ void ssm_scan(const float* __restrict__ xc, const float* __restrict__ bc,
                         const float* __restrict__ delta, const float* __restrict__ A_log,
                         const float* __restrict__ Dp, const float* __restrict__ xz,
                         float* __restrict__ g, int di) {
  int idx = blockIdx.x * blockDim.x + threadIdx.x;
  if (idx >= BATCH * di) return;
  const int d = idx % di;
  const int b = idx / di;
  float a[NSTATE], h[NSTATE];
#pragma unroll
  for (int n = 0; n < NSTATE; ++n) { a[n] = -__expf(A_log[d * NSTATE + n]); h[n] = 0.f; }
  const float dpv = Dp[d];
  for (int l = 0; l < LSEQ; ++l) {
    const int row = b * LSEQ + l;
    const float dl = delta[(size_t)row * di + d];
    const float xv = xc[(size_t)row * di + d];
    const float dx = dl * xv;
    const float* bcrow = bc + (size_t)row * 32;
    float y = 0.f;
#pragma unroll
    for (int n = 0; n < NSTATE; ++n) {
      h[n] = __expf(dl * a[n]) * h[n] + dx * bcrow[n];
      y += h[n] * bcrow[16 + n];
    }
    const float zv = xz[(size_t)row * (2 * di) + di + d];
    g[(size_t)row * di + d] = (y + dpv * xv) * silu_f(zv);
  }
}

// patch embed: (4,3,128,128) conv4x4 s4 -> tokens (4096,64), + bias + pos
__global__ void patch_embed(const float* __restrict__ x, const float* __restrict__ pw,
                            const float* __restrict__ pb, const float* __restrict__ pos,
                            float* __restrict__ t) {
  int idx = blockIdx.x * blockDim.x + threadIdx.x;
  if (idx >= BL * 64) return;
  const int o   = idx & 63;
  const int tok = idx >> 6;
  const int b   = tok >> 10;
  const int l   = tok & (LSEQ - 1);
  const int gh = l >> 5, gw = l & 31;
  float acc = pb[o];
#pragma unroll
  for (int c = 0; c < 3; ++c)
#pragma unroll
    for (int ph = 0; ph < 4; ++ph)
#pragma unroll
      for (int pq = 0; pq < 4; ++pq)
        acc += x[((size_t)(b * 3 + c) * 128 + gh * 4 + ph) * 128 + gw * 4 + pq] *
               pw[((size_t)(o * 3 + c) * 4 + ph) * 4 + pq];
  t[(size_t)tok * 64 + o] = acc + pos[(size_t)l * 64 + o];
}

__global__ void mean_tokens(const float* __restrict__ t, float* __restrict__ feat) {
  int idx = blockIdx.x * blockDim.x + threadIdx.x;
  if (idx >= BATCH * 512) return;
  const int c = idx & 511, b = idx >> 9;
  const float* p = t + (size_t)b * LSEQ * 512 + c;
  float s = 0.f;
  for (int l = 0; l < LSEQ; ++l) s += p[(size_t)l * 512];
  feat[idx] = s * (1.f / (float)LSEQ);
}

__global__ void head_kernel(const float* __restrict__ f, const float* __restrict__ w,
                            const float* __restrict__ b, float* __restrict__ out) {
  int idx = blockIdx.x * blockDim.x + threadIdx.x;
  if (idx >= BATCH * 100) return;
  const int n = idx % 100, bb = idx / 100;
  float s = b[n];
  for (int c = 0; c < 512; ++c) s += f[(size_t)bb * 512 + c] * w[(size_t)c * 100 + n];
  out[idx] = s;
}

// ---------------------------------------------------------------------------
// Host orchestration
// ---------------------------------------------------------------------------
static void launch_gemm(int epi, const float* A, const float* B, float* D,
                        const float* bias, const float* R,
                        int N, int K, int lda, int ldb, int ldd, int colbase, int ldr,
                        hipStream_t s) {
  dim3 blk(256);
  if (N % 64 == 0) {
    dim3 g(BL / 128, N / 64);
    switch (epi) {
      case EPI_NONE:
        gemm_wmma<EPI_NONE, 4><<<g, blk, 0, s>>>(A, B, D, bias, R, N, K, lda, ldb, ldd, colbase, ldr);
        break;
      case EPI_BIAS:
        gemm_wmma<EPI_BIAS, 4><<<g, blk, 0, s>>>(A, B, D, bias, R, N, K, lda, ldb, ldd, colbase, ldr);
        break;
      case EPI_BIAS_SP:
        gemm_wmma<EPI_BIAS_SP, 4><<<g, blk, 0, s>>>(A, B, D, bias, R, N, K, lda, ldb, ldd, colbase, ldr);
        break;
      default:
        gemm_wmma<EPI_BIAS_RES, 4><<<g, blk, 0, s>>>(A, B, D, bias, R, N, K, lda, ldb, ldd, colbase, ldr);
        break;
    }
  } else {                       // N multiple of 32 (the di->32 B/C projection)
    dim3 g(BL / 128, N / 32);
    switch (epi) {
      case EPI_NONE:
        gemm_wmma<EPI_NONE, 2><<<g, blk, 0, s>>>(A, B, D, bias, R, N, K, lda, ldb, ldd, colbase, ldr);
        break;
      case EPI_BIAS:
        gemm_wmma<EPI_BIAS, 2><<<g, blk, 0, s>>>(A, B, D, bias, R, N, K, lda, ldb, ldd, colbase, ldr);
        break;
      case EPI_BIAS_SP:
        gemm_wmma<EPI_BIAS_SP, 2><<<g, blk, 0, s>>>(A, B, D, bias, R, N, K, lda, ldb, ldd, colbase, ldr);
        break;
      default:
        gemm_wmma<EPI_BIAS_RES, 2><<<g, blk, 0, s>>>(A, B, D, bias, R, N, K, lda, ldb, ldd, colbase, ldr);
        break;
    }
  }
}

struct SSMp { const float *in_w, *conv_w, *conv_b, *xp_w, *dt_w, *dt_b, *A_log, *D, *out_w; };
struct Blkp { const float *nw, *nb; SSMp s[4]; const float *mw, *mb; };
struct Downp { const float *nw, *nb, *w, *b; };

extern "C" void kernel_launch(void* const* d_in, const int* in_sizes, int n_in,
                              void* d_out, int out_size, void* d_ws, size_t ws_size,
                              hipStream_t stream) {
  (void)in_sizes; (void)n_in; (void)out_size; (void)ws_size;
  int p = 0;
  auto next = [&]() { return (const float*)d_in[p++]; };

  // setup_inputs() insertion order: x, then params leaves depth-first.
  const float* x       = next();
  const float* patch_w = next();
  const float* patch_b = next();
  const float* pos     = next();

  const int depths[3] = {2, 2, 4};
  const int dims[4]   = {64, 128, 256, 512};
  Blkp blk[8];
  int bi = 0;
  for (int st = 0; st < 3; ++st)
    for (int d = 0; d < depths[st]; ++d) {
      Blkp& B = blk[bi++];
      B.nw = next(); B.nb = next();
      for (int k = 0; k < 4; ++k) {
        SSMp& S = B.s[k];
        S.in_w = next(); S.conv_w = next(); S.conv_b = next(); S.xp_w = next();
        S.dt_w = next(); S.dt_b = next();  S.A_log = next();  S.D = next();
        S.out_w = next();
      }
      B.mw = next(); B.mb = next();
    }
  Downp down[3];
  for (int i = 0; i < 3; ++i) { down[i].nw = next(); down[i].nb = next(); down[i].w = next(); down[i].b = next(); }
  const float* fnw = next();
  const float* fnb = next();
  const float* hw  = next();
  const float* hb  = next();

  // workspace carve (floats)
  float* wsp = (float*)d_ws;
  size_t off = 0;
  auto alloc = [&](size_t n) { float* r = wsp + off; off += n; return r; };
  float* bufA  = alloc((size_t)BL * 512);
  float* bufB  = alloc((size_t)BL * 512);
  float* xn_h  = alloc((size_t)BL * 256);
  float* xn_v  = alloc((size_t)BL * 256);
  float* xz    = alloc((size_t)BL * 1024);
  float* xc    = alloc((size_t)BL * 512);
  float* bcb   = alloc((size_t)BL * 32);
  float* dlt   = alloc((size_t)BL * 512);
  float* gbuf  = alloc((size_t)BL * 512);
  float* mrg   = alloc((size_t)BL * 1024);
  float* feat  = alloc((size_t)BATCH * 512);
  float* featn = alloc((size_t)BATCH * 512);

  float* cur = bufA;
  float* alt = bufB;

  patch_embed<<<(BL * 64 + 255) / 256, 256, 0, stream>>>(x, patch_w, patch_b, pos, cur);

  bi = 0;
  for (int st = 0; st < 3; ++st) {
    const int C = dims[st], di = 2 * C;
    for (int dd = 0; dd < depths[st]; ++dd) {
      Blkp& Bk = blk[bi++];
      layernorm_rows<<<(BL * 32 + 255) / 256, 256, 0, stream>>>(cur, Bk.nw, Bk.nb, xn_h, BL, C);
      transpose_tok<<<(BL * C + 255) / 256, 256, 0, stream>>>(xn_h, xn_v, C);
      for (int k = 0; k < 4; ++k) {
        SSMp& S = Bk.s[k];
        const float* seq = (k == 1) ? xn_v : xn_h;
        // xz = seq @ in_w           (4096,C)x(C,2di)
        launch_gemm(EPI_NONE, seq, S.in_w, xz, nullptr, nullptr,
                    2 * di, C, C, 2 * di, 2 * di, 0, 0, stream);
        // xc = silu(dwconv(xz[:, :di]) + cb)
        dwconv_silu<<<(BL * di + 255) / 256, 256, 0, stream>>>(xz, S.conv_w, S.conv_b, xc, di);
        // bc = xc @ xp_w            (4096,di)x(di,32)
        launch_gemm(EPI_NONE, xc, S.xp_w, bcb, nullptr, nullptr,
                    32, di, di, 32, 32, 0, 0, stream);
        // delta = softplus(xc @ dt_w + dt_b)
        launch_gemm(EPI_BIAS_SP, xc, S.dt_w, dlt, S.dt_b, nullptr,
                    di, di, di, di, di, 0, 0, stream);
        // sequential state scan + D*xc + silu(z) gate
        ssm_scan<<<(BATCH * di + 255) / 256, 256, 0, stream>>>(xc, bcb, dlt, S.A_log, S.D, xz, gbuf, di);
        // branch output into merged[:, k*C : (k+1)*C]
        launch_gemm(EPI_NONE, gbuf, S.out_w, mrg, nullptr, nullptr,
                    C, di, di, C, 4 * C, k * C, 0, stream);
      }
      // cur = cur + merged @ merge_w + merge_b   (in-place elementwise-safe)
      launch_gemm(EPI_BIAS_RES, mrg, Bk.mw, cur, Bk.mb, cur,
                  C, 4 * C, 4 * C, C, C, 0, C, stream);
    }
    // downsample: LN then linear C -> dims[st+1]
    layernorm_rows<<<(BL * 32 + 255) / 256, 256, 0, stream>>>(cur, down[st].nw, down[st].nb, xn_h, BL, C);
    launch_gemm(EPI_BIAS, xn_h, down[st].w, alt, down[st].b, nullptr,
                dims[st + 1], C, C, dims[st + 1], dims[st + 1], 0, 0, stream);
    float* t2 = cur; cur = alt; alt = t2;
  }

  mean_tokens<<<(BATCH * 512 + 255) / 256, 256, 0, stream>>>(cur, feat);
  layernorm_rows<<<1, 256, 0, stream>>>(feat, fnw, fnb, featn, BATCH, 512);
  head_kernel<<<(BATCH * 100 + 255) / 256, 256, 0, stream>>>(featn, hw, hb, (float*)d_out);
}
